// FullyAttentionalBlock_67903432950050
// MI455X (gfx1250) — matrix operand order repaired and checked by reading, not compiled, not run on hardware
//
#include <hip/hip_runtime.h>
#include <hip/hip_bf16.h>

// MI455X (gfx1250) implementation.
// Key insight: jnp.tile(enc,(1,1,w)) makes each 512-wide energy row only 4
// distinct values -> softmax collapses to a 4-way softmax and aug = (q@S)/128.
// Attention becomes memory-bound; the 3x3 conv (4.8 GFLOP) is the real compute
// and is done as an implicit GEMM on v_wmma_f32_16x16x32_bf16, with the
// weight tile staged to LDS via gfx1250 async DMA (global_load_async_to_lds).

typedef __attribute__((ext_vector_type(16))) __bf16 v16bf;
typedef __attribute__((ext_vector_type(8)))  float  v8f;

#define Bdim 4
#define Cdim 64
#define Hdim 128
#define Wdim 128
#define HWv  (Hdim*Wdim)          // 16384
#define NHWC_TOT (Bdim*Hdim*Wdim*Cdim)  // 4194304

// ---------------- prep: weight transpose to bf16 [tap][co][ci], zero BN sums
__global__ __launch_bounds__(256) void k_prepw(const float* __restrict__ Wc,
                                               __bf16* __restrict__ wct,
                                               float* __restrict__ csum,
                                               float* __restrict__ css) {
  int idx = blockIdx.x * 256 + threadIdx.x;
  if (idx < 128) { if (idx < 64) csum[idx] = 0.f; else css[idx - 64] = 0.f; }
  if (idx < 9 * 64 * 64) {
    int ci = idx & 63;
    int co = (idx >> 6) & 63;
    int tap = idx >> 12;
    wct[idx] = (__bf16)Wc[(co * 64 + ci) * 9 + tap];
  }
}

// ---------------- x (NCHW) -> out (NHWC) ; attention adds into this
__global__ __launch_bounds__(256) void k_x2nhwc(const float* __restrict__ x,
                                                float* __restrict__ out) {
  int o = blockIdx.x * 256 + threadIdx.x;       // NHWC linear
  int c = o & 63, w = (o >> 6) & 127, h = (o >> 13) & 127, b = o >> 20;
  out[o] = x[((b * 64 + c) * 128 + h) * 128 + w];
}

// ---------------- per-(b,c) plane: row means (over W) and col means (over H)
__global__ __launch_bounds__(256) void k_means(const float* __restrict__ x,
                                               float* __restrict__ mw,
                                               float* __restrict__ mh) {
  extern __shared__ char smem[];
  float* pl = (float*)smem;                     // [128][129] padded
  int bc = blockIdx.x;                          // b*64+c
  const float* xp = x + (size_t)bc * HWv;
  for (int e = threadIdx.x; e < HWv; e += 256) {
    int hh = e >> 7, ww = e & 127;
    pl[hh * 129 + ww] = xp[e];
  }
  __syncthreads();
  int t = threadIdx.x;
  if (t < 128) {
    float a = 0.f;
    for (int ww = 0; ww < 128; ++ww) a += pl[t * 129 + ww];
    mw[bc * 128 + t] = a * (1.f / 128.f);
  } else {
    int ww = t - 128;
    float a = 0.f;
    for (int hh = 0; hh < 128; ++hh) a += pl[hh * 129 + ww];
    mh[bc * 128 + ww] = a * (1.f / 128.f);
  }
}

// ---------------- enc_h[o,h,b] = sum_i W1[o,i]*mean_w[b,i,h] + b1[o]  (both paths)
__global__ __launch_bounds__(256) void k_enc(const float* __restrict__ mw,
                                             const float* __restrict__ mh,
                                             const float* __restrict__ W1,
                                             const float* __restrict__ b1,
                                             const float* __restrict__ W2,
                                             const float* __restrict__ b2,
                                             float* __restrict__ ench,
                                             float* __restrict__ encw) {
  int idx = blockIdx.x * 256 + threadIdx.x;     // 2*32768
  int p = idx >> 15;
  int r = idx & 32767;
  int o = r >> 9;
  int h = (r >> 2) & 127;
  int b = r & 3;
  const float* M  = p ? mh : mw;
  const float* Wt = p ? W2 : W1;
  const float* bs = p ? b2 : b1;
  float acc = 0.f;
  for (int i2 = 0; i2 < 64; ++i2)
    acc += Wt[o * 64 + i2] * M[(b * 64 + i2) * 128 + h];
  (p ? encw : ench)[o * 512 + h * 4 + b] = acc + bs[o];
}

// swizzled LDS index for the A tile: row pad 132, XOR-swizzle by j/64 kills
// the 8-way bank conflict of the transpose store while keeping 8-elem frag
// reads contiguous.
__device__ __forceinline__ int aidx(int j, int k) {
  return j * 132 + (k ^ (((j >> 6) & 7) << 4));
}

// ---------------- attention (PATH 0 = h-path, 1 = w-path); one block per slice i
template <int PATH>
__global__ __launch_bounds__(256) void k_attn(const float* __restrict__ x,
                                              const float* __restrict__ enc,
                                              float* __restrict__ out_nhwc,
                                              const float* __restrict__ gamma_p) {
  extern __shared__ char smem[];
  __bf16* Als = (__bf16*)smem;                       // 512 x 132 bf16 = 135168B
  float*  Sls = (float*)(smem + 512 * 132 * 2);      // [4][128]
  float*  Gls = Sls + 4 * 128;                       // [512][4]
  const int i = blockIdx.x;
  const int b = i >> 4;
  const int p0 = (i & 15) * 8;                       // w0 (PATH0) / h0 (PATH1)
  const int tid = threadIdx.x;
  const float* xb = x + (size_t)b * (Cdim * HWv);

  // Stage 1: gather A_i = feat slice into LDS (bf16)
  if (PATH == 0) {
    for (int e = tid; e < 512 * 128; e += 256) {
      int wofs = e & 7;
      int k = (e >> 3) & 127;
      int c = e >> 10;
      float v = xb[c * HWv + k * 128 + p0 + wofs];
      Als[aidx(wofs * 64 + c, k)] = (__bf16)v;
    }
  } else {
    for (int e = tid; e < 512 * 128; e += 256) {
      int k = e & 127;
      int c = (e >> 7) & 63;
      int hofs = e >> 13;
      float v = xb[c * HWv + (p0 + hofs) * 128 + k];
      Als[aidx(hofs * 64 + c, k)] = (__bf16)v;
    }
  }
  __syncthreads();

  // Stage 1b: S[t][k] = sum_{m%4==t} A[m][k]
  for (int s = tid; s < 512; s += 256) {
    int t = s & 3, k = s >> 2;
    float acc = 0.f;
    for (int mm = 0; mm < 128; ++mm)
      acc += (float)Als[aidx(mm * 4 + t, k)];
    Sls[t * 128 + k] = acc;
  }

  // Stage 2: G = A @ E via WMMA (E padded [128 x 16], cols 0..3 = enc[i,k,b'])
  const int wave = tid >> 5;
  const int lane = tid & 31;
  const int col = lane & 15;
  const int kb1 = (lane < 16) ? 0 : 16;   // A frag: elems 0..7  -> K kb1+e
  const int kb2 = kb1 + 8;                //         elems 8..15 -> K kb2+e
  const int kbB = (lane < 16) ? 0 : 16;   // B frag: elems 0..15 -> K kbB+e
  v16bf Bf[4];
#pragma unroll
  for (int kk = 0; kk < 4; ++kk) {
    v16bf f;
#pragma unroll
    for (int e = 0; e < 16; ++e) {
      int K = kk * 32 + kbB + e;
      float v = (col < 4) ? enc[i * 512 + K * 4 + col] : 0.f;
      f[e] = (__bf16)v;
    }
    Bf[kk] = f;
  }
  for (int mt = wave; mt < 32; mt += 8) {
    int row = mt * 16 + col;
    v8f acc = {};
#pragma unroll
    for (int kk = 0; kk < 4; ++kk) {
      v16bf Af;
      int o1 = kk * 32 + kb1, o2 = kk * 32 + kb2;
#pragma unroll
      for (int e = 0; e < 8; ++e) {
        Af[e]     = Als[aidx(row, o1 + e)];
        Af[8 + e] = Als[aidx(row, o2 + e)];
      }
      acc = __builtin_amdgcn_wmma_f32_16x16x32_bf16(false, Af, false, Bf[kk],
                                                    (short)0, acc, false, false);
    }
    if (col < 4) {
      int jb = mt * 16 + ((lane < 16) ? 0 : 8);
#pragma unroll
      for (int r = 0; r < 8; ++r) Gls[(jb + r) * 4 + col] = acc[r];
    }
  }
  __syncthreads();

  // Stage 3: 4-way softmax per row (multiplicity-128 cancels)
  for (int j = tid; j < 512; j += 256) {
    float g0 = Gls[j * 4 + 0], g1 = Gls[j * 4 + 1];
    float g2 = Gls[j * 4 + 2], g3 = Gls[j * 4 + 3];
    float m = fmaxf(fmaxf(g0, g1), fmaxf(g2, g3));
    float e0 = __expf(g0 - m), e1 = __expf(g1 - m);
    float e2 = __expf(g2 - m), e3 = __expf(g3 - m);
    float inv = 1.f / (e0 + e1 + e2 + e3);
    Gls[j * 4 + 0] = e0 * inv; Gls[j * 4 + 1] = e1 * inv;
    Gls[j * 4 + 2] = e2 * inv; Gls[j * 4 + 3] = e3 * inv;
  }
  __syncthreads();

  // Stage 4: out += (gamma/128) * (q @ S), coalesced NHWC writes
  const float gs = gamma_p[0] * (1.0f / 128.0f);
  const int c = tid & 63;
  const int wq = tid >> 6;
#pragma unroll
  for (int half = 0; half < 2; ++half) {
    int ofs = wq + half * 4;                 // j/64
    int j = ofs * 64 + c;
    float q0 = Gls[j * 4 + 0], q1 = Gls[j * 4 + 1];
    float q2 = Gls[j * 4 + 2], q3 = Gls[j * 4 + 3];
    for (int k = 0; k < 128; ++k) {
      float aug = q0 * Sls[0 * 128 + k] + q1 * Sls[1 * 128 + k] +
                  q2 * Sls[2 * 128 + k] + q3 * Sls[3 * 128 + k];
      int hh = (PATH == 0) ? k : (p0 + ofs);
      int ww = (PATH == 0) ? (p0 + ofs) : k;
      out_nhwc[((b * 128 + hh) * 128 + ww) * 64 + c] += gs * aug;
    }
  }
}

// ---------------- 3x3 conv as implicit GEMM (bf16 WMMA)
// Block = 64 pixels x 64 out-channels. All 72KB of [tap][co][ci] weights are
// DMA'd to LDS with gfx1250 async loads (no VGPR round-trip), overlapping the
// fp32->bf16 halo staging; completion via s_wait_asynccnt + barrier. Each wave
// owns one M-tile and two N-tiles sharing the A fragment. kh/kw are constant-
// trip nested loops so the whole 36-WMMA GEMM unrolls straight-line.
__global__ __launch_bounds__(256) void k_conv(const float* __restrict__ inp,
                                              const __bf16* __restrict__ wct,
                                              float* __restrict__ y,
                                              float* __restrict__ csum,
                                              float* __restrict__ css) {
  extern __shared__ char smem[];
  __bf16* Wl = (__bf16*)smem;                    // [9][64co][64ci]  73728B
  __bf16* Il = (__bf16*)(smem + 73728);          // [3][66][64]      25344B
  float*  Rl = (float*)(smem + 73728 + 25344);   // [128] (sum|sumsq) 512B
  const int w0 = blockIdx.x * 64;
  const int h  = blockIdx.y;
  const int n  = blockIdx.z;
  const int tid = threadIdx.x;

  // async bulk DMA: 72KB contiguous bf16 weights -> LDS (18 x b128 per lane)
  {
    const unsigned long long wgl = (unsigned long long)wct;
    unsigned gofs = tid * 16u;
#pragma unroll
    for (int it = 0; it < 18; ++it, gofs += 4096u) {
      asm volatile("global_load_async_to_lds_b128 %0, %1, %2"
                   :: "v"(gofs), "v"(gofs), "s"(wgl) : "memory");
    }
  }

  // meanwhile: stage 3x66x64 input halo, converting fp32 -> bf16
  for (int e = tid; e < 3 * 66 * 64; e += 256) {
    int ci = e & 63;
    int cc = (e >> 6) % 66;
    int r  = e / (66 * 64);
    int hh = h + r - 1;
    int ww = w0 + cc - 1;
    float v = (hh >= 0 && hh < 128 && ww >= 0 && ww < 128)
                  ? inp[((n * 128 + hh) * 128 + ww) * 64 + ci] : 0.f;
    Il[(r * 66 + cc) * 64 + ci] = (__bf16)v;
  }
  if (tid < 128) Rl[tid] = 0.f;
  asm volatile("s_wait_asynccnt 0x0" ::: "memory");
  __syncthreads();

  const int wave = tid >> 5;
  const int lane = tid & 31;
  const int col  = lane & 15;
  const int mt  = wave & 3;          // M tile (16 pixels)
  const int ntb = (wave >> 2) * 2;   // first of two N tiles
  const int kb1 = (lane < 16) ? 0 : 16;
  const int kbB = (lane < 16) ? 0 : 16;

  v8f acc0 = {}, acc1 = {};
#pragma unroll
  for (int kh = 0; kh < 3; ++kh) {
#pragma unroll
    for (int kw = 0; kw < 3; ++kw) {
      const int tap = kh * 3 + kw;
      const int cc = mt * 16 + col + kw;       // input column for my A row
#pragma unroll
      for (int kc = 0; kc < 2; ++kc) {
        v16bf Af, Bf0, Bf1;
        const __bf16* arow = &Il[(kh * 66 + cc) * 64 + kc * 32];
#pragma unroll
        for (int e = 0; e < 8; ++e) {
          Af[e]     = arow[kb1 + e];
          Af[8 + e] = arow[kb1 + 8 + e];
        }
        const __bf16* b0 = &Wl[(tap * 64 + ntb * 16 + col) * 64 + kc * 32 + kbB];
#pragma unroll
        for (int e = 0; e < 16; ++e) { Bf0[e] = b0[e]; Bf1[e] = b0[16 * 64 + e]; }
        acc0 = __builtin_amdgcn_wmma_f32_16x16x32_bf16(false, Af, false, Bf0,
                                                       (short)0, acc0, false, false);
        acc1 = __builtin_amdgcn_wmma_f32_16x16x32_bf16(false, Af, false, Bf1,
                                                       (short)0, acc1, false, false);
      }
    }
  }

  // epilogue: write pre-BN y (NCHW), accumulate channel sum / sumsq
  const int rowbase = (lane < 16) ? 0 : 8;
  const int wbase = w0 + mt * 16 + rowbase;
#pragma unroll
  for (int t2 = 0; t2 < 2; ++t2) {
    v8f a = t2 ? acc1 : acc0;
    int co = (ntb + t2) * 16 + col;
    float s = 0.f, ss = 0.f;
    size_t ybase = ((size_t)(n * 64 + co) * 128 + h) * 128 + wbase;
#pragma unroll
    for (int r = 0; r < 8; ++r) {
      float v = a[r];
      y[ybase + r] = v;
      s += v;
      ss += v * v;
    }
    atomicAdd(&Rl[co], s);
    atomicAdd(&Rl[64 + co], ss);
  }
  __syncthreads();
  if (tid < 64)       atomicAdd(&csum[tid], Rl[tid]);
  else if (tid < 128) atomicAdd(&css[tid - 64], Rl[tid]);
}

// ---------------- training-mode BN + ReLU in place on d_out
__global__ __launch_bounds__(256) void k_bn(float* __restrict__ y,
                                            const float* __restrict__ csum,
                                            const float* __restrict__ css,
                                            const float* __restrict__ bnw,
                                            const float* __restrict__ bnb) {
  int idx = blockIdx.x * 256 + threadIdx.x;
  int c = (idx >> 14) & 63;
  const float invN = 1.f / 65536.f;
  float mean = csum[c] * invN;
  float var = css[c] * invN - mean * mean;
  float v = (y[idx] - mean) * rsqrtf(var + 1e-5f) * bnw[c] + bnb[c];
  y[idx] = fmaxf(v, 0.f);
}

extern "C" void kernel_launch(void* const* d_in, const int* in_sizes, int n_in,
                              void* d_out, int out_size, void* d_ws, size_t ws_size,
                              hipStream_t stream) {
  const float* x   = (const float*)d_in[0];
  const float* W1  = (const float*)d_in[1];
  const float* b1  = (const float*)d_in[2];
  const float* W2  = (const float*)d_in[3];
  const float* b2  = (const float*)d_in[4];
  const float* Wc  = (const float*)d_in[5];
  const float* bnw = (const float*)d_in[6];
  const float* bnb = (const float*)d_in[7];
  const float* gam = (const float*)d_in[8];
  float* y = (float*)d_out;

  float* ws = (float*)d_ws;
  float* out_nhwc = ws;                        // 4194304 f
  float* mw   = out_nhwc + NHWC_TOT;           // 32768 f  [b][c][h]
  float* mh   = mw + 32768;                    // 32768 f  [b][c][w]
  float* ench = mh + 32768;                    // 32768 f  [o][h][b]
  float* encw = ench + 32768;                  // 32768 f  [o][w][b]
  float* csum = encw + 32768;                  // 64 f
  float* css  = csum + 64;                     // 64 f
  __bf16* wct = (__bf16*)(css + 64);           // 36864 bf16 [tap][co][ci]

  k_prepw<<<144, 256, 0, stream>>>(Wc, wct, csum, css);
  k_x2nhwc<<<NHWC_TOT / 256, 256, 0, stream>>>(x, out_nhwc);
  k_means<<<256, 256, 128 * 129 * 4, stream>>>(x, mw, mh);
  k_enc<<<256, 256, 0, stream>>>(mw, mh, W1, b1, W2, b2, ench, encw);
  size_t attn_lds = 512 * 132 * 2 + (4 * 128 + 512 * 4) * 4;   // 145408 B
  k_attn<0><<<64, 256, attn_lds, stream>>>(x, ench, out_nhwc, gam);
  k_attn<1><<<64, 256, attn_lds, stream>>>(x, encw, out_nhwc, gam);
  k_conv<<<dim3(2, 128, 4), 256, 73728 + 25344 + 512, stream>>>(out_nhwc, wct, y, csum, css);
  k_bn<<<NHWC_TOT / 256, 256, 0, stream>>>(y, csum, css, bnw, bnb);
}
